// TransformerBlock_19026705121964
// MI455X (gfx1250) — matrix operand order repaired and checked by reading, not compiled, not run on hardware
//
#include <hip/hip_runtime.h>
#include <hip/hip_bf16.h>
#include <cstdint>
#include <cstddef>

// ---------------------------------------------------------------------------
// Transformer block for MI455X (gfx1250, wave32, WMMA).
// All GEMMs use v_wmma_f32_16x16x32_bf16 (f32 accum). Residual stream fp32.
// GEMM weights are staged block-cooperatively into LDS with CDNA5 async
// global->LDS loads (double-buffered, s_wait_asynccnt + one barrier/iter).
// ---------------------------------------------------------------------------

#define TOKENS   65536      // B*S = 256*256
#define EMBED    384
#define HEADS    6
#define HEAD_DIM 64
#define FFDIM    1536
#define SEQ      256
#define QKV_N    1152       // 3*EMBED
#define SCORE_SCALE 0.05103103630798288f   // 384^-0.5 (embed-dim scaling, per ref)

typedef __bf16 bf16_t;
typedef __attribute__((ext_vector_type(16))) __bf16 v16bf;
typedef __attribute__((ext_vector_type(8)))  __bf16 v8bf;
typedef __attribute__((ext_vector_type(8)))  float  v8f;

static __device__ __forceinline__ bf16_t to_bf16(float f) { return (bf16_t)f; }

static __device__ __forceinline__ v8f vzero8() {
    v8f v = {0.f, 0.f, 0.f, 0.f, 0.f, 0.f, 0.f, 0.f};
    return v;
}

static __device__ __forceinline__ v16bf cat8(v8bf lo, v8bf hi) {
    return __builtin_shufflevector(lo, hi, 0,1,2,3,4,5,6,7,8,9,10,11,12,13,14,15);
}

static __device__ __forceinline__ v8f wmma_bf16(v16bf a, v16bf b, v8f c) {
    // (neg_a, A, neg_b, B, c_mod, C, reuse_a, reuse_b)
    return __builtin_amdgcn_wmma_f32_16x16x32_bf16(false, a, false, b, (short)0, c,
                                                   false, false);
}

// A-fragment (16x32 bf16) from row-major activation matrix.
// lane<16: row = rowBase+lane,    K = {kBase..+7, kBase+16..+23}
// lane>=16: row = rowBase+lane-16, K = {kBase+8..+15, kBase+24..+31}
static __device__ __forceinline__ v16bf load_afrag(const bf16_t* A, int ld,
                                                   int rowBase, int kBase) {
    int lane = threadIdx.x & 31;
    const bf16_t* p = A + (size_t)(rowBase + (lane & 15)) * ld
                        + kBase + ((lane >> 4) << 3);
    v8bf lo = *(const v8bf*)p;
    v8bf hi = *(const v8bf*)(p + 16);
    return cat8(lo, hi);
}

// B-fragment (32x16 bf16) from a packed 512-element tile (frag-order layout:
// slot l holds 16 contiguous bf16). Works for global or LDS pointers.
static __device__ __forceinline__ v16bf load_bfrag(const bf16_t* tile) {
    int lane = threadIdx.x & 31;
    const bf16_t* p = tile + lane * 16;
    v8bf lo = *(const v8bf*)p;
    v8bf hi = *(const v8bf*)(p + 8);
    return cat8(lo, hi);
}

// ---------------------------------------------------------------------------
// Weight repack: f32 [K,N] (or transposed source) -> packed bf16 B-frag tiles.
// ---------------------------------------------------------------------------
template<bool TRANSPOSED_SRC>
__global__ void k_repack_w(const float* __restrict__ W, bf16_t* __restrict__ Bp,
                           int K, int N) {
    int ntiles = N >> 4;
    int tile = blockIdx.x;
    int kt = tile / ntiles, nt = tile % ntiles;
    for (int e = threadIdx.x; e < 512; e += 256) {
        int l = e >> 4, i = e & 15;
        int k = (kt << 5) + ((i >> 3) << 4) + ((l >> 4) << 3) + (i & 7);
        int n = (nt << 4) + (l & 15);
        float v = TRANSPOSED_SRC ? W[(size_t)n * K + k] : W[(size_t)k * N + n];
        Bp[(size_t)tile * 512 + e] = to_bf16(v);
    }
}

// Fused QKV weight repack; Wq pre-scaled by 384^-0.5.
__global__ void k_repack_qkv(const float* __restrict__ Wq,
                             const float* __restrict__ Wk,
                             const float* __restrict__ Wv,
                             bf16_t* __restrict__ Bp) {
    const int ntiles = QKV_N >> 4;  // 72
    int tile = blockIdx.x;
    int kt = tile / ntiles, nt = tile % ntiles;
    for (int e = threadIdx.x; e < 512; e += 256) {
        int l = e >> 4, i = e & 15;
        int d = (kt << 5) + ((i >> 3) << 4) + ((l >> 4) << 3) + (i & 7);     // reduction index
        int n = (nt << 4) + (l & 15);                // output column
        const float* src = (n < 384) ? Wq : (n < 768) ? Wk : Wv;
        int m = (n < 384) ? n : (n < 768) ? n - 384 : n - 768;
        int h = m >> 6, kk = m & 63;
        float v = src[((size_t)h * EMBED + d) * HEAD_DIM + kk];
        if (n < 384) v *= SCORE_SCALE;
        Bp[(size_t)tile * 512 + e] = to_bf16(v);
    }
}

// ---------------------------------------------------------------------------
// LayerNorm: one wave per 384-wide row, 8 rows per 256-thread block.
// ---------------------------------------------------------------------------
__global__ void k_layernorm(const float* __restrict__ x,
                            const float* __restrict__ g,
                            const float* __restrict__ b,
                            bf16_t* __restrict__ out) {
    int lane = threadIdx.x & 31;
    int row  = blockIdx.x * 8 + (threadIdx.x >> 5);
    const float* xr = x + (size_t)row * EMBED;
    float v[12];
    float s = 0.f;
#pragma unroll
    for (int i = 0; i < 12; ++i) { v[i] = xr[i * 32 + lane]; s += v[i]; }
#pragma unroll
    for (int off = 16; off >= 1; off >>= 1) s += __shfl_xor(s, off, 32);
    float mu = s * (1.f / EMBED);
    float q = 0.f;
#pragma unroll
    for (int i = 0; i < 12; ++i) { float d = v[i] - mu; q += d * d; }
#pragma unroll
    for (int off = 16; off >= 1; off >>= 1) q += __shfl_xor(q, off, 32);
    float rs = rsqrtf(q * (1.f / EMBED) + 1e-5f);
    bf16_t* outr = out + (size_t)row * EMBED;
#pragma unroll
    for (int i = 0; i < 12; ++i) {
        int c = i * 32 + lane;
        outr[c] = to_bf16((v[i] - mu) * rs * g[c] + b[c]);
    }
}

// ---------------------------------------------------------------------------
// bf16 WMMA GEMM with async-LDS weight staging.
// Block = 8 waves; wave -> 16 rows x 64 cols. The 4KB of packed B per k-step
// is shared by all 8 waves, so it is staged once per block into LDS with
// global_load_async_to_lds_b128 (double buffer; issue kt+1 after the kt
// barrier so the DMA overlaps the WMMAs).
// ---------------------------------------------------------------------------
template<bool BIAS, bool RELU, bool RESID, bool OUTBF, int KSTEPS>
__global__ void __launch_bounds__(256, 1)
k_gemm(const bf16_t* __restrict__ A,
       const bf16_t* __restrict__ Bp,
       const float* __restrict__ bias,
       const float* __restrict__ resid,
       float* __restrict__ outf,
       bf16_t* __restrict__ outb,
       int N) {
    const int K = KSTEPS * 32;
    __shared__ bf16_t bstage[2][4 * 512];   // 2 x 4KB

    int lane = threadIdx.x & 31;
    int wave = threadIdx.x >> 5;
    int rowBase = (blockIdx.x * 8 + wave) << 4;
    int nt0 = blockIdx.y << 2;
    int ntiles = N >> 4;

    // Issue one 16B async global->LDS copy per thread (4KB per stage).
    auto issueStage = [&](int sbuf, int kt) {
        const bf16_t* src = Bp + ((size_t)(kt * ntiles + nt0) << 9)
                               + threadIdx.x * 8;
        unsigned ldsOff = (unsigned)(uintptr_t)&bstage[sbuf][threadIdx.x * 8];
        unsigned long long ga = (unsigned long long)(uintptr_t)src;
        asm volatile("global_load_async_to_lds_b128 %0, %1, off"
                     :: "v"(ldsOff), "v"(ga) : "memory");
    };

    v8f acc[4];
#pragma unroll
    for (int t = 0; t < 4; ++t) acc[t] = vzero8();

    const bf16_t* aRow = A + (size_t)(rowBase + (lane & 15)) * K + ((lane >> 4) << 3);

    issueStage(0, 0);
    for (int kt = 0; kt < KSTEPS; ++kt) {
        asm volatile("s_wait_asynccnt 0x0" ::: "memory");  // stage kt resident
        __syncthreads();                                   // ...for all waves
        if (kt + 1 < KSTEPS) issueStage((kt + 1) & 1, kt + 1);

        v16bf a = load_afrag(A, K, rowBase, kt << 5);
        if (kt + 1 < KSTEPS) __builtin_prefetch(aRow + ((kt + 1) << 5), 0, 3);

        const bf16_t* sb = &bstage[kt & 1][0];
#pragma unroll
        for (int t = 0; t < 4; ++t) {
            v16bf bfr = load_bfrag(sb + (t << 9));   // ds_load_b128 x2
            acc[t] = wmma_bf16(a, bfr, acc[t]);
        }
        // next iteration's barrier protects stage kt&1 against overwrite
    }

    int rOff = (lane >> 4) << 3;
    int nloc = lane & 15;
#pragma unroll
    for (int t = 0; t < 4; ++t) {
        int col = ((nt0 + t) << 4) + nloc;
        float bv = BIAS ? bias[col] : 0.f;
#pragma unroll
        for (int r = 0; r < 8; ++r) {
            int row = rowBase + rOff + r;
            float v = acc[t][r] + bv;
            if (RELU)  v = fmaxf(v, 0.f);
            if (RESID) v += resid[(size_t)row * N + col];
            if (OUTBF) outb[(size_t)row * N + col] = to_bf16(v);
            else       outf[(size_t)row * N + col] = v;
        }
    }
}

// ---------------------------------------------------------------------------
// Repack per-(b,h) K and V into B-frag tile order (contiguous b128 loads).
// ---------------------------------------------------------------------------
__global__ void k_repack_kv(const bf16_t* __restrict__ qkv,
                            bf16_t* __restrict__ kpack,
                            bf16_t* __restrict__ vpack) {
    int bh = blockIdx.x;
    int b = bh / HEADS, h = bh % HEADS;
    size_t tok0 = (size_t)b * SEQ;
    size_t base = (size_t)bh * 16384;
    for (int idx = threadIdx.x; idx < 16384; idx += 256) {
        int tile = idx >> 9, e = idx & 511;
        int l = e >> 4, i = e & 15;
        int ktile = tile >> 1, c = tile & 1;
        int key  = (ktile << 4) + (l & 15);
        int dim  = (c << 5) + ((i >> 3) << 4) + ((l >> 4) << 3) + (i & 7);
        kpack[base + idx] = qkv[(tok0 + key) * QKV_N + EMBED + h * HEAD_DIM + dim];
        int kc = tile >> 2, nt = tile & 3;
        int key2 = (kc << 5) + ((i >> 3) << 4) + ((l >> 4) << 3) + (i & 7);
        int dim2 = (nt << 4) + (l & 15);
        vpack[base + idx] = qkv[(tok0 + key2) * QKV_N + 2 * EMBED + h * HEAD_DIM + dim2];
    }
}

// ---------------------------------------------------------------------------
// Flash attention (causal, online softmax). One wave per 16-row Q tile.
// qtile is readfirstlane'd so all control flow is provably wave-uniform
// (scalar branches, EXEC untouched around WMMA). Unmasked chunks run in a
// tight loop; exactly one masked chunk (kcFull == qrow0>>5) is peeled.
// Row sums come from a ones-column WMMA riding with the PV accumulators.
// ---------------------------------------------------------------------------
__global__ void __launch_bounds__(256, 1)
k_attn(const bf16_t* __restrict__ qkv,
       const bf16_t* __restrict__ kpack,
       const bf16_t* __restrict__ vpack,
       bf16_t* __restrict__ heads) {
    __shared__ bf16_t plds[8 * 16 * 32];   // 1 KB per wave

    int lane = threadIdx.x & 31;
    int wave = threadIdx.x >> 5;
    int bh = blockIdx.y;
    int b = bh / HEADS, h = bh % HEADS;
    int qtile = __builtin_amdgcn_readfirstlane(blockIdx.x * 8 + wave); // uniform
    int qrow0 = qtile << 4;
    size_t tok0 = (size_t)b * SEQ;

    // Q fragments (2 dim-chunks of 32), resident. (Already scaled via Wq.)
    v16bf qa[2];
#pragma unroll
    for (int c = 0; c < 2; ++c) {
        const bf16_t* p = qkv + (tok0 + qrow0 + (lane & 15)) * QKV_N
                              + h * HEAD_DIM + (c << 5) + ((lane >> 4) << 3);
        v8bf lo = *(const v8bf*)p;
        v8bf hi = *(const v8bf*)(p + 16);
        qa[c] = cat8(lo, hi);
    }

    // Ones B-fragment for the row-sum WMMA.
    v16bf ones;
#pragma unroll
    for (int i = 0; i < 16; ++i) ones[i] = (bf16_t)1.0f;

    float m[8];
#pragma unroll
    for (int r = 0; r < 8; ++r) m[r] = -__builtin_inff();
    v8f acc[4];
#pragma unroll
    for (int t = 0; t < 4; ++t) acc[t] = vzero8();
    v8f lacc = vzero8();

    const bf16_t* kbh = kpack + (size_t)bh * 16384;
    const bf16_t* vbh = vpack + (size_t)bh * 16384;
    bf16_t* pw = plds + wave * 512;

    int rOff = (lane >> 4) << 3;
    int nloc = lane & 15;
    int kcFull = qrow0 >> 5;   // scalar: final (only masked) chunk index

    auto chunk = [&](int kc, bool masked) {
        // ---- scores: 16x32 (two 16x16 tiles), 4 WMMAs ----
        v8f s0 = vzero8(), s1 = vzero8();
#pragma unroll
        for (int c = 0; c < 2; ++c) {
            v16bf kb0 = load_bfrag(kbh + ((size_t)(((kc * 2 + 0) << 1) + c) << 9));
            s0 = wmma_bf16(qa[c], kb0, s0);
            v16bf kb1 = load_bfrag(kbh + ((size_t)(((kc * 2 + 1) << 1) + c) << 9));
            s1 = wmma_bf16(qa[c], kb1, s1);
        }
        if (masked) {
            int k0 = (kc << 5) + nloc;
            int k1 = k0 + 16;
#pragma unroll
            for (int r = 0; r < 8; ++r) {
                int rw = qrow0 + rOff + r;
                s0[r] = (k0 <= rw) ? s0[r] : -__builtin_inff();
                s1[r] = (k1 <= rw) ? s1[r] : -__builtin_inff();
            }
        }
        // ---- online softmax per row ----
#pragma unroll
        for (int r = 0; r < 8; ++r) {
            float mx = fmaxf(s0[r], s1[r]);
#pragma unroll
            for (int off = 8; off >= 1; off >>= 1)
                mx = fmaxf(mx, __shfl_xor(mx, off, 32));
            float mn = fmaxf(m[r], mx);
            float alpha = __expf(m[r] - mn);
            m[r] = mn;
            float e0 = __expf(s0[r] - mn);
            float e1 = __expf(s1[r] - mn);
#pragma unroll
            for (int t = 0; t < 4; ++t) acc[t][r] *= alpha;
            lacc[r] *= alpha;
            int prow = rOff + r;
            pw[prow * 32 + nloc]      = to_bf16(e0);
            pw[prow * 32 + 16 + nloc] = to_bf16(e1);
        }
        // ---- P (A-layout from LDS) x V, plus ones-column row-sum WMMA ----
        const bf16_t* pp = pw + (lane & 15) * 32 + ((lane >> 4) << 3);
        v8bf plo = *(const v8bf*)pp;
        v8bf phi = *(const v8bf*)(pp + 16);
        v16bf pa = cat8(plo, phi);
#pragma unroll
        for (int t = 0; t < 4; ++t) {
            v16bf vb = load_bfrag(vbh + ((size_t)((kc << 2) + t) << 9));
            acc[t] = wmma_bf16(pa, vb, acc[t]);
        }
        lacc = wmma_bf16(pa, ones, lacc);
    };

    for (int kc = 0; kc < kcFull; ++kc) chunk(kc, false);
    chunk(kcFull, true);

    // ---- normalize and write heads [T, 384], col = h*64 + dim ----
#pragma unroll
    for (int r = 0; r < 8; ++r) {
        float inv = 1.f / lacc[r];
        int row = qrow0 + rOff + r;
#pragma unroll
        for (int t = 0; t < 4; ++t) {
            int col = h * HEAD_DIM + (t << 4) + nloc;
            heads[(tok0 + row) * EMBED + col] = to_bf16(acc[t][r] * inv);
        }
    }
}

// ---------------------------------------------------------------------------
// Host-side orchestration.
// ---------------------------------------------------------------------------
extern "C" void kernel_launch(void* const* d_in, const int* in_sizes, int n_in,
                              void* d_out, int out_size, void* d_ws, size_t ws_size,
                              hipStream_t stream) {
    const float* x     = (const float*)d_in[0];
    const float* Wq    = (const float*)d_in[1];
    const float* Wk    = (const float*)d_in[2];
    const float* Wv    = (const float*)d_in[3];
    const float* Wproj = (const float*)d_in[4];
    const float* bproj = (const float*)d_in[5];
    const float* W1    = (const float*)d_in[6];
    const float* b1    = (const float*)d_in[7];
    const float* W2    = (const float*)d_in[8];
    const float* b2    = (const float*)d_in[9];
    const float* g1    = (const float*)d_in[10];
    const float* be1   = (const float*)d_in[11];
    const float* g2    = (const float*)d_in[12];
    const float* be2   = (const float*)d_in[13];

    char* p = (char*)d_ws;
    auto carve = [&](size_t bytes) {
        void* r = (void*)p;
        p += (bytes + 255) & ~(size_t)255;
        return r;
    };
    bf16_t* xn     = (bf16_t*)carve((size_t)TOKENS * EMBED * 2);
    bf16_t* qkv    = (bf16_t*)carve((size_t)TOKENS * QKV_N * 2);
    bf16_t* kpack  = (bf16_t*)carve((size_t)TOKENS * EMBED * 2);
    bf16_t* vpack  = (bf16_t*)carve((size_t)TOKENS * EMBED * 2);
    bf16_t* headsb = (bf16_t*)carve((size_t)TOKENS * EMBED * 2);
    float*  xmid   = (float*) carve((size_t)TOKENS * EMBED * 4);
    bf16_t* hbuf   = (bf16_t*)carve((size_t)TOKENS * FFDIM * 2);
    bf16_t* wqkvp  = (bf16_t*)carve((size_t)EMBED * QKV_N * 2);
    bf16_t* wprojp = (bf16_t*)carve((size_t)EMBED * EMBED * 2);
    bf16_t* w1p    = (bf16_t*)carve((size_t)EMBED * FFDIM * 2);
    bf16_t* w2p    = (bf16_t*)carve((size_t)FFDIM * EMBED * 2);

    // Weight repacks (tiny; run once per launch).
    k_repack_qkv<<<dim3((EMBED / 32) * (QKV_N / 16)), 256, 0, stream>>>(Wq, Wk, Wv, wqkvp);
    k_repack_w<true ><<<dim3((EMBED / 32) * (EMBED / 16)), 256, 0, stream>>>(Wproj, wprojp, EMBED, EMBED);
    k_repack_w<false><<<dim3((EMBED / 32) * (FFDIM / 16)), 256, 0, stream>>>(W1, w1p, EMBED, FFDIM);
    k_repack_w<false><<<dim3((FFDIM / 32) * (EMBED / 16)), 256, 0, stream>>>(W2, w2p, FFDIM, EMBED);

    // LN1 -> xn (bf16)
    k_layernorm<<<dim3(TOKENS / 8), 256, 0, stream>>>(x, g1, be1, xn);

    // QKV: [T,384] x [384,1152] -> qkv bf16 (Q pre-scaled by 384^-0.5)
    k_gemm<false, false, false, true, EMBED / 32>
        <<<dim3(TOKENS / 128, QKV_N / 64), 256, 0, stream>>>(
        xn, wqkvp, nullptr, nullptr, nullptr, qkv, QKV_N);

    // K/V frag repack per (b,h)
    k_repack_kv<<<dim3((TOKENS / SEQ) * HEADS), 256, 0, stream>>>(qkv, kpack, vpack);

    // Flash attention -> heads bf16 [T,384]
    k_attn<<<dim3(SEQ / 128, (TOKENS / SEQ) * HEADS), 256, 0, stream>>>(qkv, kpack, vpack, headsb);

    // Projection + bias + residual(x) -> xmid f32
    k_gemm<true, false, true, false, EMBED / 32>
        <<<dim3(TOKENS / 128, EMBED / 64), 256, 0, stream>>>(
        headsb, wprojp, bproj, x, xmid, nullptr, EMBED);

    // LN2 -> xn (reuse)
    k_layernorm<<<dim3(TOKENS / 8), 256, 0, stream>>>(xmid, g2, be2, xn);

    // FFN1 + bias + ReLU -> hbuf bf16
    k_gemm<true, true, false, true, EMBED / 32>
        <<<dim3(TOKENS / 128, FFDIM / 64), 256, 0, stream>>>(
        xn, w1p, b1, nullptr, nullptr, hbuf, FFDIM);

    // FFN2 + bias + residual(xmid) -> d_out f32
    k_gemm<true, false, true, false, FFDIM / 32>
        <<<dim3(TOKENS / 128, EMBED / 64), 256, 0, stream>>>(
        hbuf, w2p, b2, xmid, (float*)d_out, nullptr, EMBED);
}
